// CustomGRUCell_28389733827226
// MI455X (gfx1250) — compile-verified, matrix operand order checked by reading, not run on hardware
//
#include <hip/hip_runtime.h>
#include <stdint.h>

typedef __attribute__((ext_vector_type(16))) __bf16        v16bf;
typedef __attribute__((ext_vector_type(8)))  float         v8f;
typedef __attribute__((ext_vector_type(8)))  unsigned int  v8u;

union Frag {
    v8u   u;
    v16bf b;
};

__device__ __forceinline__ unsigned short f32_to_bf16(float f) {
    unsigned int x = __float_as_uint(f);
    unsigned int r = x + 0x7FFFu + ((x >> 16) & 1u);   // round-to-nearest-even
    return (unsigned short)(r >> 16);
}

// K index held by (half, j) for 16-bit WMMA operands (pairs k, k+1).
// lanes 0-15: V0..V3 -> K=0..7, V4..V7 -> K=16..23 ; lanes 16-31: +8
__device__ __forceinline__ int kmap(int half, int j) {
    int kb = half * 8;
    return (j < 4) ? (2 * j + kb) : (16 + 2 * (j - 4) + kb);
}

__device__ __forceinline__ v8f wmma_bf16(const Frag& a, const Frag& b, v8f c) {
    return __builtin_amdgcn_wmma_f32_16x16x32_bf16(
        /*neg_a=*/false, a.b, /*neg_b=*/false, b.b,
        /*c_mod=*/(short)0, c, /*reuse_a=*/false, /*reuse_b=*/false);
}

// ---------------------------------------------------------------------------
// Pre-pass: pack one 256x256 f32 row-major weight matrix into bf16 B-fragment
// layout: out[((nt*8 + kc)*32 + lane)*8 + j] = pack(W[k][n], W[k+1][n])
//   n = nt*16 + (lane&15),  k = kc*32 + kmap(lane>>4, j)
// 32768 dwords per matrix; one thread per output dword.
// ---------------------------------------------------------------------------
__global__ __launch_bounds__(256) void pack_weight_kernel(
    const float* __restrict__ W, uint32_t* __restrict__ out)
{
    int gid  = blockIdx.x * blockDim.x + threadIdx.x;   // 0..32767
    int j    = gid & 7;
    int t    = gid >> 3;
    int lane = t & 31;
    int kc   = (t >> 5) & 7;
    int nt   = t >> 8;
    int n    = nt * 16 + (lane & 15);
    int k    = kc * 32 + kmap(lane >> 4, j);
    unsigned short lo = f32_to_bf16(W[k * 256 + n]);
    unsigned short hi = f32_to_bf16(W[(k + 1) * 256 + n]);
    out[gid] = (uint32_t)lo | ((uint32_t)hi << 16);
}

// ---------------------------------------------------------------------------
// Fragment loads
// ---------------------------------------------------------------------------
__device__ __forceinline__ Frag load_a_lds(const unsigned short (*lds)[256],
                                           int mbase, int kc, int lane)
{
    Frag f;
    int half = lane >> 4;
    int m    = mbase + (lane & 15);
    const uint32_t* p = (const uint32_t*)&lds[m][kc * 32];
#pragma unroll
    for (int j = 0; j < 8; ++j)
        f.u[j] = p[kmap(half, j) >> 1];
    return f;
}

__device__ __forceinline__ Frag load_b_ws(const uint32_t* __restrict__ w,
                                          int nt, int kc, int lane)
{
    Frag f;
    f.u = *(const v8u*)(w + ((((nt * 8 + kc) * 32) + lane) << 3));
    return f;
}

// ---------------------------------------------------------------------------
// Fused GRU cell. Block = 256 threads (8 waves) = 32 rows of B.
// Wave w owns n-tiles {2w, 2w+1} x m-tiles {0,1}  (4 16x16 positions).
// ---------------------------------------------------------------------------
__global__ __launch_bounds__(256) void gru_fused_kernel(
    const float* __restrict__ x, const float* __restrict__ h,
    const uint32_t* __restrict__ wWz, const uint32_t* __restrict__ wUz,
    const uint32_t* __restrict__ wWr, const uint32_t* __restrict__ wUr,
    const uint32_t* __restrict__ wWh, const uint32_t* __restrict__ wUh,
    const float* __restrict__ bz, const float* __restrict__ br,
    const float* __restrict__ bh, float* __restrict__ out)
{
    __shared__ unsigned short x_lds[32][256];
    __shared__ unsigned short h_lds[32][256];
    __shared__ unsigned short rh_lds[32][256];

    const int tid  = threadIdx.x;
    const int wave = tid >> 5;
    const int lane = tid & 31;
    const int half = lane >> 4;
    const int l15  = lane & 15;
    const int row0 = blockIdx.x * 32;

    // Stage x, h tiles into LDS as bf16 (coalesced f32 reads).
    for (int idx = tid; idx < 32 * 256; idx += 256) {
        int r = idx >> 8, c = idx & 255;
        x_lds[r][c] = f32_to_bf16(x[(row0 + r) * 256 + c]);
        h_lds[r][c] = f32_to_bf16(h[(row0 + r) * 256 + c]);
    }
    __syncthreads();

    const int nt0 = wave * 2;
    float bzv[2], brv[2], bhv[2];
#pragma unroll
    for (int ntl = 0; ntl < 2; ++ntl) {
        int n = (nt0 + ntl) * 16 + l15;
        bzv[ntl] = bz[n]; brv[ntl] = br[n]; bhv[ntl] = bh[n];
    }

    v8f accZ[2][2] = {};   // [mt][ntl]  z pre-activation, later holds z
    v8f accR[2][2] = {};   //            r pre-activation
    v8f accH[2][2] = {};   //            xWh, later += rh*Uh

    // -------- Phase A: preZ = x Wz + h Uz ; preR = x Wr + h Ur ; XH = x Wh
    for (int kc = 0; kc < 8; ++kc) {
        Frag ax0 = load_a_lds(x_lds,  0, kc, lane);
        Frag ax1 = load_a_lds(x_lds, 16, kc, lane);
        Frag ah0 = load_a_lds(h_lds,  0, kc, lane);
        Frag ah1 = load_a_lds(h_lds, 16, kc, lane);
#pragma unroll
        for (int ntl = 0; ntl < 2; ++ntl) {
            const int nt = nt0 + ntl;
            Frag bWz = load_b_ws(wWz, nt, kc, lane);
            accZ[0][ntl] = wmma_bf16(ax0, bWz, accZ[0][ntl]);
            accZ[1][ntl] = wmma_bf16(ax1, bWz, accZ[1][ntl]);
            Frag bUz = load_b_ws(wUz, nt, kc, lane);
            accZ[0][ntl] = wmma_bf16(ah0, bUz, accZ[0][ntl]);
            accZ[1][ntl] = wmma_bf16(ah1, bUz, accZ[1][ntl]);
            Frag bWr = load_b_ws(wWr, nt, kc, lane);
            accR[0][ntl] = wmma_bf16(ax0, bWr, accR[0][ntl]);
            accR[1][ntl] = wmma_bf16(ax1, bWr, accR[1][ntl]);
            Frag bUr = load_b_ws(wUr, nt, kc, lane);
            accR[0][ntl] = wmma_bf16(ah0, bUr, accR[0][ntl]);
            accR[1][ntl] = wmma_bf16(ah1, bUr, accR[1][ntl]);
            Frag bWh = load_b_ws(wWh, nt, kc, lane);
            accH[0][ntl] = wmma_bf16(ax0, bWh, accH[0][ntl]);
            accH[1][ntl] = wmma_bf16(ax1, bWh, accH[1][ntl]);
        }
    }

    // -------- Gates: z, r ; write rh = r * h_prev (bf16) into LDS
#pragma unroll
    for (int mt = 0; mt < 2; ++mt) {
#pragma unroll
        for (int ntl = 0; ntl < 2; ++ntl) {
            const int n = (nt0 + ntl) * 16 + l15;
#pragma unroll
            for (int v = 0; v < 8; ++v) {
                const int ml   = v + half * 8;         // C layout: M = v (+8 for hi half)
                const int grow = row0 + mt * 16 + ml;
                float zv = 1.0f / (1.0f + __expf(-(accZ[mt][ntl][v] + bzv[ntl])));
                float rv = 1.0f / (1.0f + __expf(-(accR[mt][ntl][v] + brv[ntl])));
                float hp = h[grow * 256 + n];
                rh_lds[mt * 16 + ml][n] = f32_to_bf16(rv * hp);
                accZ[mt][ntl][v] = zv;                 // keep z in regs
            }
        }
    }
    __syncthreads();

    // -------- Phase B: XH += rh @ Uh
    for (int kc = 0; kc < 8; ++kc) {
        Frag a0 = load_a_lds(rh_lds,  0, kc, lane);
        Frag a1 = load_a_lds(rh_lds, 16, kc, lane);
#pragma unroll
        for (int ntl = 0; ntl < 2; ++ntl) {
            Frag bUh = load_b_ws(wUh, nt0 + ntl, kc, lane);
            accH[0][ntl] = wmma_bf16(a0, bUh, accH[0][ntl]);
            accH[1][ntl] = wmma_bf16(a1, bUh, accH[1][ntl]);
        }
    }

    // -------- h = z*h_prev + (1-z)*tanh(XH + bh)
#pragma unroll
    for (int mt = 0; mt < 2; ++mt) {
#pragma unroll
        for (int ntl = 0; ntl < 2; ++ntl) {
            const int n = (nt0 + ntl) * 16 + l15;
#pragma unroll
            for (int v = 0; v < 8; ++v) {
                const int ml   = v + half * 8;
                const int grow = row0 + mt * 16 + ml;
                float hhat = tanhf(accH[mt][ntl][v] + bhv[ntl]);
                float zv   = accZ[mt][ntl][v];
                float hp   = h[grow * 256 + n];
                out[grow * 256 + n] = zv * hp + (1.0f - zv) * hhat;
            }
        }
    }
}

// ---------------------------------------------------------------------------
extern "C" void kernel_launch(void* const* d_in, const int* in_sizes, int n_in,
                              void* d_out, int out_size, void* d_ws, size_t ws_size,
                              hipStream_t stream)
{
    const float* x  = (const float*)d_in[0];
    const float* h  = (const float*)d_in[1];
    const float* Wz = (const float*)d_in[2];
    const float* Uz = (const float*)d_in[3];
    const float* bz = (const float*)d_in[4];
    const float* Wr = (const float*)d_in[5];
    const float* Ur = (const float*)d_in[6];
    const float* br = (const float*)d_in[7];
    const float* Wh = (const float*)d_in[8];
    const float* Uh = (const float*)d_in[9];
    const float* bh = (const float*)d_in[10];

    uint32_t* ws = (uint32_t*)d_ws;
    const int MD = 256 * 256 / 2;   // 32768 dwords per packed bf16 matrix

    // Pack 6 weight matrices into WMMA-B fragment layout (bf16) in d_ws.
    pack_weight_kernel<<<128, 256, 0, stream>>>(Wz, ws + 0 * MD);
    pack_weight_kernel<<<128, 256, 0, stream>>>(Uz, ws + 1 * MD);
    pack_weight_kernel<<<128, 256, 0, stream>>>(Wr, ws + 2 * MD);
    pack_weight_kernel<<<128, 256, 0, stream>>>(Ur, ws + 3 * MD);
    pack_weight_kernel<<<128, 256, 0, stream>>>(Wh, ws + 4 * MD);
    pack_weight_kernel<<<128, 256, 0, stream>>>(Uh, ws + 5 * MD);

    const int B = 65536;
    gru_fused_kernel<<<B / 32, 256, 0, stream>>>(
        x, h,
        ws + 0 * MD, ws + 1 * MD, ws + 2 * MD,
        ws + 3 * MD, ws + 4 * MD, ws + 5 * MD,
        bz, br, bh, (float*)d_out);
}
